// AttentionPooling_59983513256114
// MI455X (gfx1250) — matrix operand order repaired and checked by reading, not compiled, not run on hardware
//
#include <hip/hip_runtime.h>
#include <hip/hip_bf16.h>

typedef __attribute__((ext_vector_type(16))) __bf16 v16bf;
typedef __attribute__((ext_vector_type(8)))  float  v8f;

#define DD 256
#define AH 256

// Branch-free tanh: native V_TANH_F32 if the builtin exists, else a
// divergence-free composition of fast exp + fast divide (single trans ops).
__device__ __forceinline__ float fast_tanh(float v) {
#if __has_builtin(__builtin_amdgcn_tanhf)
  return __builtin_amdgcn_tanhf(v);
#else
  float e = __expf(2.f * v);             // v_exp_f32 (scaled), no branches
  return 1.f - __fdividef(2.f, e + 1.f); // v_rcp_f32 + mul, no branches
#endif
}

// ---------------------------------------------------------------------------
// Kernel 1: swizzle W1 (fp32 [256,256] row-major, [k][n]) into bf16 in the
// exact per-lane WMMA B-matrix order so the GEMM does contiguous 32B loads.
// Output element index = ((ct*8 + kt)*32 + lane)*16 + e
//   n = ct*16 + (lane&15),  K = kt*32 + 8*(lane>>4) + (e&7) + 16*(e>>3)  (ISA B 32x16)
// ---------------------------------------------------------------------------
__global__ void prep_w1_kernel(const float* __restrict__ W1, __bf16* __restrict__ w1s) {
  int idx  = blockIdx.x * 256 + threadIdx.x;      // 65536 total
  int e    = idx & 15;
  int lane = (idx >> 4) & 31;
  int kt   = (idx >> 9) & 7;
  int ct   = idx >> 12;
  int n    = ct * 16 + (lane & 15);
  int k    = kt * 32 + 8 * (lane >> 4) + (e & 7) + 16 * (e >> 3);
  w1s[idx] = (__bf16)W1[k * DD + n];
}

// ---------------------------------------------------------------------------
// Kernel 2: scores[n] = tanh(x[n]@W1 + b1) @ W2 + b2 via bf16 WMMA.
// Wave handles 16 rows. A tiles (8 x v16bf) cached in VGPRs. B tiles staged
// through ping-pong LDS (2 x 8KB): next tile's global loads are issued
// before the current tile's WMMAs so the memory latency overlaps compute.
// ---------------------------------------------------------------------------
__global__ __launch_bounds__(256) void score_kernel(
    const float*  __restrict__ x,
    const __bf16* __restrict__ w1s,
    const float*  __restrict__ b1,
    const float*  __restrict__ W2,
    const float*  __restrict__ b2,
    float*        __restrict__ scores,
    int N) {
  __shared__ __align__(32) __bf16 lb[2 * 8 * 32 * 16];   // 16 KB ping-pong

  const int tid  = threadIdx.x;
  const int lane = tid & 31;
  const int wave = tid >> 5;
  const int half = lane >> 4;
  const int col  = lane & 15;

  const long rowBase = (long)blockIdx.x * 128 + (long)wave * 16;
  long r0 = rowBase + col;            // A layout: lanes L and L+16 share row m=L&15
  if (r0 >= N) r0 = N - 1;            // clamp (keeps EXEC uniform for WMMA)

  // ---- load & convert A tiles: 16 rows x 256 K, bf16, ISA 16-bit A layout ----
  v16bf a[8];
  const float* xrow = x + r0 * DD;
#pragma unroll
  for (int kt = 0; kt < 8; ++kt) {
    const float* px = xrow + kt * 32 + 8 * half;   // e=0..7  -> K = kt*32 + 8*half + e
    float t[16];
#pragma unroll
    for (int j = 0; j < 8; ++j) t[j] = px[j];
#pragma unroll
    for (int j = 0; j < 8; ++j) t[8 + j] = px[16 + j];  // e=8..15 -> +16
#pragma unroll
    for (int j = 0; j < 16; ++j) a[kt][j] = (__bf16)t[j];
  }

  float sp[8];
#pragma unroll
  for (int r = 0; r < 8; ++r) sp[r] = 0.f;

  const v16bf* lbv  = (const v16bf*)lb;
  const uint4* gsrc = (const uint4*)w1s;   // tile ct starts at gsrc + ct*512
  uint4*       lbu  = (uint4*)lb;

  // ---- prestage tile 0 ----
  {
    uint4 s0 = gsrc[tid * 2];
    uint4 s1 = gsrc[tid * 2 + 1];
    lbu[tid * 2]     = s0;
    lbu[tid * 2 + 1] = s1;
  }
  __syncthreads();

  // ---- 16 column tiles of 16; bias folded into C init ----
  for (int ct = 0; ct < 16; ++ct) {
    // issue next tile's global loads now; they complete during the WMMAs
    uint4 n0, n1;
    if (ct < 15) {
      n0 = gsrc[(ct + 1) * 512 + tid * 2];
      n1 = gsrc[(ct + 1) * 512 + tid * 2 + 1];
    }

    const v16bf* cur = lbv + (ct & 1) * 256;   // 256 v16bf per 8KB buffer

    const float bias = b1[ct * 16 + col];
    v8f acc;
#pragma unroll
    for (int r = 0; r < 8; ++r) acc[r] = bias;

#pragma unroll
    for (int kt = 0; kt < 8; ++kt) {
      v16bf bt = cur[kt * 32 + lane];
      acc = __builtin_amdgcn_wmma_f32_16x16x32_bf16(
          /*neg_a=*/false, a[kt], /*neg_b=*/false, bt,
          /*c_mod=*/(short)0, acc, /*reuse_a=*/false, /*reuse_b=*/false);
    }

    const float w2v = W2[ct * 16 + col];
#pragma unroll
    for (int r = 0; r < 8; ++r) sp[r] += fast_tanh(acc[r]) * w2v;

    // commit staged tile into the other buffer, then one barrier
    if (ct < 15) {
      uint4* dst = lbu + ((ct + 1) & 1) * 512;
      dst[tid * 2]     = n0;
      dst[tid * 2 + 1] = n1;
      __syncthreads();
    }
  }

  // ---- reduce over the 16 columns (within each half-wave) ----
#pragma unroll
  for (int r = 0; r < 8; ++r) {
    float v = sp[r];
    v += __shfl_xor(v, 8, 16);
    v += __shfl_xor(v, 4, 16);
    v += __shfl_xor(v, 2, 16);
    v += __shfl_xor(v, 1, 16);
    sp[r] = v;
  }

  const float bb = b2[0];
  if (col == 0) {                      // lanes 0 (rows r) and 16 (rows r+8)
#pragma unroll
    for (int r = 0; r < 8; ++r) {
      long row = rowBase + r + 8 * half;
      if (row < N) scores[row] = sp[r] + bb;
    }
  }
}

// ---------------------------------------------------------------------------
// Kernel 3: segment bounds via binary search on the sorted batch array.
// segstart[b] = lower_bound(batch, b); segstart[B] = N. Empty segs handled.
// ---------------------------------------------------------------------------
__global__ void seg_bounds_kernel(const int* __restrict__ batch,
                                  int* __restrict__ segstart, int N, int B) {
  int b = blockIdx.x * 256 + threadIdx.x;
  if (b > B) return;
  if (b == B) { segstart[B] = N; return; }
  int lo = 0, hi = N;
  while (lo < hi) {
    int mid = (lo + hi) >> 1;
    if (batch[mid] < b) lo = mid + 1; else hi = mid;
  }
  segstart[b] = lo;
}

// ---------------------------------------------------------------------------
// Kernel 4: per-segment online softmax stats (max + sum of exp), 1 wave/seg.
// ---------------------------------------------------------------------------
__global__ __launch_bounds__(256) void seg_stats_kernel(
    const float* __restrict__ scores, const int* __restrict__ segstart,
    float* __restrict__ segmax, float* __restrict__ seginv, int B) {
  int wave = threadIdx.x >> 5, lane = threadIdx.x & 31;
  int b = blockIdx.x * 8 + wave;
  if (b >= B) return;
  int s = segstart[b], e = segstart[b + 1];

  float m = -__builtin_inff();
  float sum = 0.f;
  for (int i = s + lane; i < e; i += 32) {
    float v  = scores[i];
    float nm = fmaxf(m, v);
    sum = sum * __expf(m - nm) + __expf(v - nm);   // m=-inf -> factor 0, safe
    m = nm;
  }
#pragma unroll
  for (int off = 16; off > 0; off >>= 1) {
    float om = __shfl_xor(m,   off, 32);
    float os = __shfl_xor(sum, off, 32);
    float nm = fmaxf(m, om);
    float fa = (m  > -__builtin_inff()) ? __expf(m  - nm) : 0.f;
    float fb = (om > -__builtin_inff()) ? __expf(om - nm) : 0.f;
    sum = sum * fa + os * fb;
    m = nm;
  }
  if (lane == 0) {
    segmax[b] = m;
    seginv[b] = (e > s) ? 1.f / sum : 0.f;
  }
}

// ---------------------------------------------------------------------------
// Kernel 5: normalized weights w[i] = exp(s - max) * inv_denom.
// ---------------------------------------------------------------------------
__global__ void weights_kernel(const float* __restrict__ scores,
                               const int* __restrict__ batch,
                               const float* __restrict__ segmax,
                               const float* __restrict__ seginv,
                               float* __restrict__ w, int N) {
  int i = blockIdx.x * 256 + threadIdx.x;
  if (i >= N) return;
  int b = batch[i];
  w[i] = __expf(scores[i] - segmax[b]) * seginv[b];
}

// ---------------------------------------------------------------------------
// Kernel 6: pooled[b, :] = sum_{i in seg b} x[i, :] * w[i].
// One block per segment; thread d owns column d; rows are contiguous.
// ---------------------------------------------------------------------------
__global__ __launch_bounds__(256) void pool_kernel(
    const float* __restrict__ x, const float* __restrict__ w,
    const int* __restrict__ segstart, float* __restrict__ out) {
  int b = blockIdx.x;
  int d = threadIdx.x;
  int s = segstart[b], e = segstart[b + 1];
  float acc = 0.f;
  for (int i = s; i < e; ++i) {
    __builtin_prefetch(x + (long)(i + 8) * DD + d, 0, 0);  // global_prefetch_b8
    acc += x[(long)i * DD + d] * w[i];
  }
  out[(long)b * DD + d] = acc;
}

// ---------------------------------------------------------------------------
extern "C" void kernel_launch(void* const* d_in, const int* in_sizes, int n_in,
                              void* d_out, int out_size, void* d_ws, size_t ws_size,
                              hipStream_t stream) {
  const float* x     = (const float*)d_in[0];
  const int*   batch = (const int*)  d_in[1];
  const float* W1    = (const float*)d_in[2];
  const float* b1    = (const float*)d_in[3];
  const float* W2    = (const float*)d_in[4];
  const float* b2    = (const float*)d_in[5];
  float* out = (float*)d_out;

  const int N = in_sizes[0] / DD;
  const int B = out_size / DD;

  // workspace layout
  char* ws = (char*)d_ws;
  __bf16* w1s     = (__bf16*)ws;                               // 131072 B
  float*  scores  = (float*)(ws + 131072);                     // N*4
  float*  wbuf    = (float*)(ws + 131072 + (size_t)N * 4);     // N*4
  char*   p       = ws + 131072 + 2 * (size_t)N * 4;
  int*    segstart= (int*)p;                                   // (B+1)*4
  float*  segmax  = (float*)(p + (size_t)(B + 1) * 4);         // B*4
  float*  seginv  = (float*)(p + (size_t)(B + 1) * 4 + (size_t)B * 4);

  prep_w1_kernel<<<(DD * AH) / 256, 256, 0, stream>>>(W1, w1s);

  score_kernel<<<(N + 127) / 128, 256, 0, stream>>>(x, w1s, b1, W2, b2, scores, N);

  seg_bounds_kernel<<<(B + 1 + 255) / 256, 256, 0, stream>>>(batch, segstart, N, B);

  seg_stats_kernel<<<(B + 7) / 8, 256, 0, stream>>>(scores, segstart, segmax, seginv, B);

  weights_kernel<<<(N + 255) / 256, 256, 0, stream>>>(scores, batch, segmax, seginv, wbuf, N);

  pool_kernel<<<B, 256, 0, stream>>>(x, wbuf, segstart, out);
}
